// SABlock_31026843746722
// MI455X (gfx1250) — compile-verified, matrix operand order checked by reading, not run on hardware
//
#include <hip/hip_runtime.h>
#include <hip/hip_bf16.h>

typedef __attribute__((ext_vector_type(8)))  _Float16 v8h;
typedef __attribute__((ext_vector_type(16))) _Float16 v16h;
typedef __attribute__((ext_vector_type(8)))  float    v8f;
typedef __attribute__((ext_vector_type(4)))  int      v4i;

static constexpr int Bc = 4;
static constexpr int Sc = 2048;
static constexpr int Dc = 768;
static constexpr int Hc = 12;
static constexpr int HDc = 64;           // head dim
static constexpr float SCALEc = 0.125f;  // 64^-0.5

#if defined(__has_builtin)
#if __has_builtin(__builtin_amdgcn_global_load_async_to_lds_b128)
#define USE_ASYNC_LDS 1
#endif
#endif
#ifndef USE_ASYNC_LDS
#define USE_ASYNC_LDS 0
#endif

__device__ __forceinline__ v16h pack2(v8h lo, v8h hi) {
  return __builtin_shufflevector(lo, hi, 0,1,2,3,4,5,6,7,8,9,10,11,12,13,14,15);
}
// Load a 16-bit A/B fragment half-pair: 8 halves at p, 8 halves at p+16
__device__ __forceinline__ v16h load_frag16(const _Float16* p) {
  return pack2(*(const v8h*)p, *(const v8h*)(p + 16));
}
__device__ __forceinline__ v8f wmma16(v16h a, v16h b, v8f c) {
  return __builtin_amdgcn_wmma_f32_16x16x32_f16(false, a, false, b, (short)0, c, false, false);
}
__device__ __forceinline__ float halfwave_max(float v) {
#pragma unroll
  for (int off = 1; off < 16; off <<= 1) v = fmaxf(v, __shfl_xor(v, off, 32));
  return v;
}
__device__ __forceinline__ float halfwave_sum(float v) {
#pragma unroll
  for (int off = 1; off < 16; off <<= 1) v += __shfl_xor(v, off, 32);
  return v;
}

// 16-byte global -> LDS copy (async path on gfx1250: global_load_async_to_lds_b128)
__device__ __forceinline__ void copy16_g2l(_Float16* lp, const _Float16* gp) {
#if USE_ASYNC_LDS
  __builtin_amdgcn_global_load_async_to_lds_b128(
      (__attribute__((address_space(1))) v4i*)(void*)gp,
      (__attribute__((address_space(3))) v4i*)(void*)lp, 0, 0);
#else
  *(v8h*)lp = *(const v8h*)gp;
#endif
}
__device__ __forceinline__ void wait_async_copies() {
#if USE_ASYNC_LDS
#if __has_builtin(__builtin_amdgcn_s_wait_asynccnt)
  __builtin_amdgcn_s_wait_asynccnt(0);
#else
  asm volatile("s_wait_asynccnt 0x0" ::: "memory");
#endif
#endif
}

// ---------------- elementwise f32 -> f16 convert ----------------
__global__ void convert_f32_f16(const float* __restrict__ src,
                                _Float16* __restrict__ dst, int n) {
  int idx = blockIdx.x * blockDim.x + threadIdx.x;
  if (idx < n) dst[idx] = (_Float16)src[idx];
}

// ---------------- weight transpose + f32->f16 convert ----------------
__global__ void transpose_f32_f16(const float* __restrict__ src,
                                  _Float16* __restrict__ dst,
                                  int rows, int cols) {
  int idx = blockIdx.x * blockDim.x + threadIdx.x;
  if (idx < rows * cols) {
    int r = idx / cols, c = idx % cols;
    dst[(size_t)c * rows + r] = (_Float16)src[idx];
  }
}

// ---------------- QKV projection: [B*S,D] @ [D,3D] ----------------
// One wave computes a 16x64 tile (4 accumulators, A-fragment reused 4x).
// wT is w_qkv transposed: [3D][D] f16. Outputs scattered into
// q[B,H,S,64], k[B,H,S,64], vT[B,H,64,S] (V pre-transposed for PV).
__global__ __launch_bounds__(32) void qkv_gemm(const _Float16* __restrict__ xh,
                                               const _Float16* __restrict__ wT,
                                               _Float16* __restrict__ q,
                                               _Float16* __restrict__ k,
                                               _Float16* __restrict__ vT) {
  const int m0   = blockIdx.x * 16;        // row in [0, B*S)
  const int n0   = blockIdx.y * 64;        // col in [0, 3*D), 64-aligned
  const int lane = threadIdx.x;
  const int rsel = lane & 15;
  const int koff = (lane >> 4) * 8;

  v8f acc[4] = {};
  for (int kb = 0; kb < Dc; kb += 32) {
    v16h a = load_frag16(xh + (size_t)(m0 + rsel) * Dc + kb + koff);
#pragma unroll
    for (int j = 0; j < 4; ++j) {
      v16h b = load_frag16(wT + (size_t)(n0 + j * 16 + rsel) * Dc + kb + koff);
      acc[j] = wmma16(a, b, acc[j]);
    }
  }

  const int which = n0 / Dc;              // 0=q 1=k 2=v (64-col tile fits a head)
  const int h     = (n0 % Dc) / HDc;
#pragma unroll
  for (int j = 0; j < 4; ++j) {
    const int hd = j * 16 + rsel;
#pragma unroll
    for (int r = 0; r < 8; ++r) {
      int m = m0 + r + ((lane >> 4) << 3);
      int b_ = m / Sc, s = m % Sc;
      _Float16 val = (_Float16)acc[j][r];
      if (which == 0)
        q[(((size_t)b_ * Hc + h) * Sc + s) * HDc + hd] = val;
      else if (which == 1)
        k[(((size_t)b_ * Hc + h) * Sc + s) * HDc + hd] = val;
      else
        vT[(((size_t)b_ * Hc + h) * HDc + hd) * Sc + s] = val;
    }
  }
}

// ---------------- flash attention ----------------
// Workgroup = 4 waves = 64 Q rows of one (b,h). K/V staged in LDS in 64-key
// blocks shared by all 4 waves; each wave owns a 16-row Q tile with online
// softmax in the WMMA C-layout (rows in VGPR index, cols in lanes).
__global__ __launch_bounds__(128) void attn_kernel(const _Float16* __restrict__ q,
                                                   const _Float16* __restrict__ k,
                                                   const _Float16* __restrict__ vT,
                                                   _Float16* __restrict__ ctx) {
  __shared__ _Float16 kTile[64 * 64];       // [key][ch]   8 KB
  __shared__ _Float16 vTile[64 * 64];       // [ch][key]   8 KB
  __shared__ _Float16 pTile[4][16 * 64];    // per-wave P staging, 2 KB each

  const int bh   = blockIdx.x;              // b*H + h
  const int wave = threadIdx.x >> 5;
  const int lane = threadIdx.x & 31;
  const int tid  = threadIdx.x;
  const int q0   = blockIdx.y * 64 + wave * 16;
  const int rsel = lane & 15;
  const int koff = (lane >> 4) * 8;
  const int rhi  = (lane >> 4) << 3;        // 0 or 8

  const _Float16* qp = q  + ((size_t)bh * Sc + q0) * HDc;
  const _Float16* kp = k  + (size_t)bh * Sc * HDc;
  const _Float16* vp = vT + (size_t)bh * HDc * Sc;

  // Q fragments (16x64 = two 16x32 A-frags), pre-scaled by 1/sqrt(HD)
  v16h qa0, qa1;
  {
    const _Float16* base = qp + (size_t)rsel * HDc + koff;
    v16h t0 = load_frag16(base);
    v16h t1 = load_frag16(base + 32);
    const _Float16 sc = (_Float16)SCALEc;
#pragma unroll
    for (int i = 0; i < 16; ++i) { t0[i] = t0[i] * sc; t1[i] = t1[i] * sc; }
    qa0 = t0; qa1 = t1;
  }

  float m_run[8], l_run[8];
#pragma unroll
  for (int r = 0; r < 8; ++r) { m_run[r] = -1e30f; l_run[r] = 0.f; }
  v8f O[4] = {};

  for (int kb = 0; kb < Sc; kb += 64) {
    // ---- cooperative staging of K (64x64) and V^T (64x64) into LDS ----
#pragma unroll
    for (int i = 0; i < 4; ++i) {
      int c   = tid + 128 * i;              // 512 16-byte chunks each
      int row = c >> 3, ofs = (c & 7) * 8;
      copy16_g2l(&kTile[row * 64 + ofs], kp + (size_t)(kb + row) * HDc + ofs);
      copy16_g2l(&vTile[row * 64 + ofs], vp + (size_t)row * Sc + kb + ofs);
    }
    if (kb + 64 < Sc) {                     // prefetch next block into L2
      __builtin_prefetch(kp + (size_t)(kb + 64 + (tid >> 1)) * HDc + (tid & 1) * 32, 0, 1);
      __builtin_prefetch(vp + (size_t)(tid >> 1) * Sc + kb + 64 + (tid & 1) * 32, 0, 1);
    }
    wait_async_copies();
    __syncthreads();

    // ---- scores: four 16x16 tiles over 64 keys, K-dim = 64 channels ----
    v8f s[4];
#pragma unroll
    for (int t = 0; t < 4; ++t) {
      const _Float16* kbase = &kTile[(t * 16 + rsel) * 64 + koff];
      v8f st = {};
      st = wmma16(qa0, load_frag16(kbase), st);
      st = wmma16(qa1, load_frag16(kbase + 32), st);
      s[t] = st;
    }

    // ---- online softmax update ----
    float alpha[8];
#pragma unroll
    for (int r = 0; r < 8; ++r) {
      float cmax = fmaxf(fmaxf(s[0][r], s[1][r]), fmaxf(s[2][r], s[3][r]));
      cmax = halfwave_max(cmax);
      float mnew = fmaxf(m_run[r], cmax);
      alpha[r]   = __expf(m_run[r] - mnew);
      float psum = 0.f;
      int prow = (r + rhi) * 64 + rsel;     // C-layout row in P staging
#pragma unroll
      for (int t = 0; t < 4; ++t) {
        float p = __expf(s[t][r] - mnew);
        psum += p;
        pTile[wave][prow + t * 16] = (_Float16)p;
      }
      l_run[r] = l_run[r] * alpha[r] + halfwave_sum(psum);
      m_run[r] = mnew;
    }
#pragma unroll
    for (int n = 0; n < 4; ++n)
#pragma unroll
      for (int r = 0; r < 8; ++r) O[n][r] *= alpha[r];

    // reload P in A-fragment layout (two 16x32 frags over 64 keys)
    const _Float16* pl = &pTile[wave][rsel * 64 + koff];
    v16h pa0 = load_frag16(pl);
    v16h pa1 = load_frag16(pl + 32);

    // ---- PV: V^T rows in LDS are contiguous along keys ----
#pragma unroll
    for (int n = 0; n < 4; ++n) {
      const _Float16* vb = &vTile[(n * 16 + rsel) * 64 + koff];
      O[n] = wmma16(pa0, load_frag16(vb), O[n]);
      O[n] = wmma16(pa1, load_frag16(vb + 32), O[n]);
    }
    __syncthreads();   // protect LDS tiles before next staging
  }

  // ---- normalize and store context [B,S,768] f16 ----
  const int b_ = bh / Hc, h = bh % Hc;
#pragma unroll
  for (int r = 0; r < 8; ++r) {
    float inv = 1.f / l_run[r];
    int s = q0 + r + rhi;
#pragma unroll
    for (int n = 0; n < 4; ++n) {
      ctx[((size_t)b_ * Sc + s) * Dc + h * HDc + n * 16 + rsel] =
          (_Float16)(O[n][r] * inv);
    }
  }
}

// ---------------- output projection: ctx @ w_out + b_out ----------------
__global__ __launch_bounds__(32) void out_gemm(const _Float16* __restrict__ ctx,
                                               const _Float16* __restrict__ wT,
                                               const float* __restrict__ bias,
                                               float* __restrict__ out) {
  const int m0   = blockIdx.x * 16;
  const int n0   = blockIdx.y * 64;
  const int lane = threadIdx.x;
  const int rsel = lane & 15;
  const int koff = (lane >> 4) * 8;

  v8f acc[4] = {};
  for (int kb = 0; kb < Dc; kb += 32) {
    v16h a = load_frag16(ctx + (size_t)(m0 + rsel) * Dc + kb + koff);
#pragma unroll
    for (int j = 0; j < 4; ++j) {
      v16h b = load_frag16(wT + (size_t)(n0 + j * 16 + rsel) * Dc + kb + koff);
      acc[j] = wmma16(a, b, acc[j]);
    }
  }
#pragma unroll
  for (int j = 0; j < 4; ++j) {
    float bv = bias[n0 + j * 16 + rsel];
#pragma unroll
    for (int r = 0; r < 8; ++r) {
      int m = m0 + r + ((lane >> 4) << 3);
      out[(size_t)m * Dc + n0 + j * 16 + rsel] = acc[j][r] + bv;
    }
  }
}

extern "C" void kernel_launch(void* const* d_in, const int* in_sizes, int n_in,
                              void* d_out, int out_size, void* d_ws, size_t ws_size,
                              hipStream_t stream) {
  const float* x     = (const float*)d_in[0];
  const float* w_qkv = (const float*)d_in[1];
  const float* w_out = (const float*)d_in[2];
  const float* b_out = (const float*)d_in[3];
  float* out = (float*)d_out;

  const size_t nqkv = (size_t)Bc * Hc * Sc * HDc;   // 6,291,456 per tensor
  _Float16* q     = (_Float16*)d_ws;
  _Float16* k     = q + nqkv;
  _Float16* vT    = k + nqkv;
  _Float16* ctx   = vT + nqkv;
  _Float16* xh    = ctx + (size_t)Bc * Sc * Dc;
  _Float16* wqkvT = xh + (size_t)Bc * Sc * Dc;
  _Float16* woutT = wqkvT + (size_t)Dc * 3 * Dc;

  {
    int nx = Bc * Sc * Dc;
    convert_f32_f16<<<(nx + 255) / 256, 256, 0, stream>>>(x, xh, nx);
    int n1 = Dc * 3 * Dc;
    transpose_f32_f16<<<(n1 + 255) / 256, 256, 0, stream>>>(w_qkv, wqkvT, Dc, 3 * Dc);
    int n2 = Dc * Dc;
    transpose_f32_f16<<<(n2 + 255) / 256, 256, 0, stream>>>(w_out, woutT, Dc, Dc);
  }
  qkv_gemm<<<dim3(Bc * Sc / 16, 3 * Dc / 64), 32, 0, stream>>>(xh, wqkvT, q, k, vT);
  attn_kernel<<<dim3(Bc * Hc, Sc / 64), 128, 0, stream>>>(q, k, vT, ctx);
  out_gemm<<<dim3(Bc * Sc / 16, Dc / 64), 32, 0, stream>>>(ctx, woutT, b_out, out);
}
